// Attention_49168785605257
// MI455X (gfx1250) — matrix operand order repaired and checked by reading, not compiled, not run on hardware
//
#include <hip/hip_runtime.h>
#include <hip/hip_bf16.h>

typedef _Float16 half_t;
typedef __attribute__((ext_vector_type(16))) _Float16 v16h;
typedef __attribute__((ext_vector_type(8)))  float    v8f;

// ---------------------------------------------------------------------------
// CDNA5 async copy: global -> LDS, 16 bytes per lane, tracked by ASYNCcnt.
// ---------------------------------------------------------------------------
__device__ __forceinline__ void async_copy_b128(void* lds_ptr, const void* gptr) {
  unsigned lds_off = (unsigned)(unsigned long long)lds_ptr;   // low 32 bits = LDS offset
  unsigned long long ga = (unsigned long long)gptr;
  asm volatile("global_load_async_to_lds_b128 %0, %1, off"
               :: "v"(lds_off), "v"(ga) : "memory");
}
__device__ __forceinline__ void wait_async0() {
  asm volatile("s_wait_asynccnt 0x0" ::: "memory");
}

// ---------------------------------------------------------------------------
// Max-reduce across the 16-lane row group using DPP16 (no LDS, no addr math).
// xor1 = quad_perm[1,0,3,2]=0xB1, xor2 = quad_perm[2,3,0,1]=0x4E,
// xor4 = ROW_HALF_MIRROR=0x141, xor8 = ROW_MIRROR=0x140.
// ---------------------------------------------------------------------------
__device__ __forceinline__ float rowmax16_dpp(float x) {
  int t;
  t = __builtin_amdgcn_update_dpp(0, __float_as_int(x), 0xB1,  0xF, 0xF, true);
  x = fmaxf(x, __int_as_float(t));
  t = __builtin_amdgcn_update_dpp(0, __float_as_int(x), 0x4E,  0xF, 0xF, true);
  x = fmaxf(x, __int_as_float(t));
  t = __builtin_amdgcn_update_dpp(0, __float_as_int(x), 0x141, 0xF, 0xF, true);
  x = fmaxf(x, __int_as_float(t));
  t = __builtin_amdgcn_update_dpp(0, __float_as_int(x), 0x140, 0xF, 0xF, true);
  x = fmaxf(x, __int_as_float(t));
  return x;
}

// ---------------------------------------------------------------------------
// WMMA fragment loaders (16x16x32 f16, wave32). All contiguous-in-K.
// ---------------------------------------------------------------------------
__device__ __forceinline__ v16h load_a_frag(const half_t* base, int ld) {
  const int lane = threadIdx.x & 31;
  const int m = lane & 15, hf = lane >> 4;
  v16h f;
#pragma unroll
  for (int i = 0; i < 8; ++i) {
    f[i]     = base[m * ld + hf * 16 + i];
    f[8 + i] = base[m * ld + hf * 16 + 8 + i];
  }
  return f;
}

// bt: [N][K]-layout tile; koff selects a 32-wide K slice
__device__ __forceinline__ v16h load_bt_frag(const half_t* bt, int ld, int koff) {
  const int lane = threadIdx.x & 31;
  const int n = lane & 15, hf = lane >> 4;
  v16h f;
#pragma unroll
  for (int j = 0; j < 16; ++j)
    f[j] = bt[n * ld + koff + hf * 16 + j];
  return f;
}

// ---------------------------------------------------------------------------
// One-time prep kernels
// ---------------------------------------------------------------------------
__global__ __launch_bounds__(256) void convert_f32_f16(
    const float* __restrict__ src, half_t* __restrict__ dst, int n) {
  int i = (blockIdx.x * 256 + threadIdx.x) * 8;
  if (i < n) {
#pragma unroll
    for (int j = 0; j < 8; ++j) dst[i + j] = (half_t)src[i + j];
  }
}

// Wt[n][k] = (f16) W[k][n]
__global__ __launch_bounds__(256) void wt_transpose(
    const float* __restrict__ W, half_t* __restrict__ Wt, int K, int N) {
  __shared__ half_t T[64][72];
  const int r0 = blockIdx.y * 64, c0 = blockIdx.x * 64;
  const int t = threadIdx.x;
  {
    const int rr = t >> 2, cc0 = (t & 3) * 16;
#pragma unroll
    for (int i = 0; i < 16; ++i)
      T[rr][cc0 + i] = (half_t)W[(size_t)(r0 + rr) * N + c0 + cc0 + i];
  }
  __syncthreads();
  {
    const int nn = t >> 2, kk0 = (t & 3) * 16;
#pragma unroll
    for (int i = 0; i < 16; ++i)
      Wt[(size_t)(c0 + nn) * K + r0 + kk0 + i] = T[kk0 + i][nn];
  }
}

// vt[(b*H+h)*64 + dh][j] = kv[(b*N+j)][512 + h*64 + dh]
__global__ __launch_bounds__(256) void v_transpose(
    const half_t* __restrict__ kv, half_t* __restrict__ vt, int Nn) {
  __shared__ half_t T[64][72];
  const int j0 = blockIdx.x * 64, h = blockIdx.y, b = blockIdx.z;
  const int t = threadIdx.x;
  const half_t* src = kv + ((size_t)(b * Nn) + j0) * 1024 + 512 + h * 64;
  {
    const int jr = t >> 2, c0 = (t & 3) * 16;
#pragma unroll
    for (int i = 0; i < 16; ++i)
      T[jr][c0 + i] = src[(size_t)jr * 1024 + c0 + i];
  }
  __syncthreads();
  half_t* dst = vt + ((size_t)(b * 8 + h) * 64) * Nn + j0;
  {
    const int dh = t >> 2, jj0 = (t & 3) * 16;
#pragma unroll
    for (int i = 0; i < 16; ++i)
      dst[(size_t)dh * Nn + jj0 + i] = T[jj0 + i][dh];
  }
}

// ---------------------------------------------------------------------------
// GEMM: C[M,N] = alpha * (A @ B) + bias, A f16 [M,K], Bt f16 [N][K].
// 8 waves, tile 128(M) x 64(N), K-step 64, async double-buffered LDS fills.
// ---------------------------------------------------------------------------
template <typename TOut, bool HAS_BIAS>
__global__ __launch_bounds__(256) void gemm_wmma(
    const half_t* __restrict__ A, const half_t* __restrict__ Bt,
    const float* __restrict__ bias, TOut* __restrict__ C,
    int M, int N, int K, float alpha)
{
  __shared__ half_t As[2][128][72];
  __shared__ half_t WsT[2][64][72];

  const int tid  = threadIdx.x;
  const int m0   = blockIdx.y * 128;
  const int n0   = blockIdx.x * 64;
  const int wave = tid >> 5;
  const int wvm  = wave & 3;
  const int wvn  = wave >> 2;

  const int ar = tid >> 1, ac = (tid & 1) * 32;   // A tile 128 x 64
  const int wr = tid >> 2, wc = (tid & 3) * 16;   // B tile 64 x 64

  auto issue = [&](int k0, int pb) {
    const half_t* as = A + (size_t)(m0 + ar) * K + k0 + ac;
    async_copy_b128(&As[pb][ar][ac],      as);
    async_copy_b128(&As[pb][ar][ac + 8],  as + 8);
    async_copy_b128(&As[pb][ar][ac + 16], as + 16);
    async_copy_b128(&As[pb][ar][ac + 24], as + 24);
    const half_t* bs = Bt + (size_t)(n0 + wr) * K + k0 + wc;
    async_copy_b128(&WsT[pb][wr][wc],     bs);
    async_copy_b128(&WsT[pb][wr][wc + 8], bs + 8);
  };

  v8f acc[2][2] = {};
  issue(0, 0);
  int pb = 0;
  for (int k0 = 0; k0 < K; k0 += 64, pb ^= 1) {
    wait_async0();
    __syncthreads();
    if (k0 + 64 < K) issue(k0 + 64, pb ^ 1);

#pragma unroll
    for (int ks = 0; ks < 2; ++ks) {
      v16h a0 = load_a_frag(&As[pb][wvm * 32][ks * 32], 72);
      v16h a1 = load_a_frag(&As[pb][wvm * 32 + 16][ks * 32], 72);
      v16h b0 = load_bt_frag(&WsT[pb][wvn * 32][0], 72, ks * 32);
      v16h b1 = load_bt_frag(&WsT[pb][wvn * 32 + 16][0], 72, ks * 32);
      acc[0][0] = __builtin_amdgcn_wmma_f32_16x16x32_f16(false, a0, false, b0,
                                                         (short)0, acc[0][0], false, false);
      acc[0][1] = __builtin_amdgcn_wmma_f32_16x16x32_f16(false, a0, false, b1,
                                                         (short)0, acc[0][1], false, false);
      acc[1][0] = __builtin_amdgcn_wmma_f32_16x16x32_f16(false, a1, false, b0,
                                                         (short)0, acc[1][0], false, false);
      acc[1][1] = __builtin_amdgcn_wmma_f32_16x16x32_f16(false, a1, false, b1,
                                                         (short)0, acc[1][1], false, false);
    }
  }

  const int lane = tid & 31;
  const int nn = lane & 15, hf = lane >> 4;
#pragma unroll
  for (int mi = 0; mi < 2; ++mi) {
#pragma unroll
    for (int ni = 0; ni < 2; ++ni) {
      const int mbase = m0 + wvm * 32 + mi * 16 + hf * 8;
      const int ncol  = n0 + wvn * 32 + ni * 16 + nn;
      const float bv  = HAS_BIAS ? bias[ncol] : 0.f;
#pragma unroll
      for (int r = 0; r < 8; ++r) {
        float v = acc[mi][ni][r] * alpha + bv;
        C[(size_t)(mbase + r) * N + ncol] = (TOut)v;
      }
    }
  }
}

// ---------------------------------------------------------------------------
// Flash attention chunk (64 keys). Row sums via WMMA against the ones-row
// appended to V^T; row max via DPP16 butterflies.
// ---------------------------------------------------------------------------
template <bool MASKED>
__device__ __forceinline__ void attn_chunk(
    const v16h* aq, float* mrow, v8f* accO, v8f& accL,
    const half_t (*Ks)[72], const half_t (*VsT)[72], half_t (*Psw)[72],
    int q0, int c)
{
  const int lane = threadIdx.x & 31;
  const int n = lane & 15, hf = lane >> 4;

  // S = Q @ K^T : four 16x16 key tiles, contraction 64
  v8f s[4];
#pragma unroll
  for (int kt = 0; kt < 4; ++kt) {
    v16h b0, b1;
#pragma unroll
    for (int j = 0; j < 16; ++j) {
      b0[j] = Ks[kt * 16 + n][hf * 16 + j];
      b1[j] = Ks[kt * 16 + n][32 + hf * 16 + j];
    }
    v8f sv = {};
    sv = __builtin_amdgcn_wmma_f32_16x16x32_f16(false, aq[0], false, b0,
                                                (short)0, sv, false, false);
    sv = __builtin_amdgcn_wmma_f32_16x16x32_f16(false, aq[1], false, b1,
                                                (short)0, sv, false, false);
    s[kt] = sv;
  }

  float corr[8];
#pragma unroll
  for (int r = 0; r < 8; ++r) {
    if (MASKED) {
      const int qi = q0 + r + 8 * hf;
#pragma unroll
      for (int kt = 0; kt < 4; ++kt)
        if (c + kt * 16 + n > qi) s[kt][r] = -3.0e38f;
    }
    float rmax = fmaxf(fmaxf(s[0][r], s[1][r]), fmaxf(s[2][r], s[3][r]));
    rmax = rowmax16_dpp(rmax);
    const float mnew = fmaxf(mrow[r], rmax);
    corr[r] = __expf(mrow[r] - mnew);
    mrow[r] = mnew;
#pragma unroll
    for (int kt = 0; kt < 4; ++kt) {
      if (MASKED)
        s[kt][r] = (s[kt][r] <= -1.0e37f) ? 0.f : __expf(s[kt][r] - mnew);
      else
        s[kt][r] = __expf(s[kt][r] - mnew);
    }
  }
#pragma unroll
  for (int t = 0; t < 4; ++t)
#pragma unroll
    for (int r = 0; r < 8; ++r)
      accO[t][r] = accO[t][r] * corr[r];
#pragma unroll
  for (int r = 0; r < 8; ++r)
    accL[r] = accL[r] * corr[r];

  // P (C layout) -> per-wave LDS -> two A-frags (16 x 64)
#pragma unroll
  for (int kt = 0; kt < 4; ++kt)
#pragma unroll
    for (int r = 0; r < 8; ++r)
      Psw[r + 8 * hf][kt * 16 + n] = (half_t)s[kt][r];
  v16h ap0, ap1;
  {
    const int m = lane & 15;
#pragma unroll
    for (int i = 0; i < 8; ++i) {
      ap0[i]     = Psw[m][hf * 16 + i];
      ap0[8 + i] = Psw[m][hf * 16 + 8 + i];
      ap1[i]     = Psw[m][32 + hf * 16 + i];
      ap1[8 + i] = Psw[m][32 + hf * 16 + 8 + i];
    }
  }

  // l += P @ ones  (ones row = VsT row 64)
  {
    v16h o0, o1;
#pragma unroll
    for (int j = 0; j < 16; ++j) {
      o0[j] = VsT[64 + n][hf * 16 + j];
      o1[j] = VsT[64 + n][32 + hf * 16 + j];
    }
    accL = __builtin_amdgcn_wmma_f32_16x16x32_f16(false, ap0, false, o0,
                                                  (short)0, accL, false, false);
    accL = __builtin_amdgcn_wmma_f32_16x16x32_f16(false, ap1, false, o1,
                                                  (short)0, accL, false, false);
  }

  // O += P @ V : 4 dh-tiles, contraction 64 keys
#pragma unroll
  for (int t = 0; t < 4; ++t) {
    v16h bv0, bv1;
#pragma unroll
    for (int j = 0; j < 16; ++j) {
      bv0[j] = VsT[t * 16 + n][hf * 16 + j];
      bv1[j] = VsT[t * 16 + n][32 + hf * 16 + j];
    }
    accO[t] = __builtin_amdgcn_wmma_f32_16x16x32_f16(false, ap0, false, bv0,
                                                     (short)0, accO[t], false, false);
    accO[t] = __builtin_amdgcn_wmma_f32_16x16x32_f16(false, ap1, false, bv1,
                                                     (short)0, accO[t], false, false);
  }
}

// ---------------------------------------------------------------------------
// Flash attention (causal). Block = 8 waves x 16 q-rows; 64-key chunks,
// async double-buffered K / V^T staging.
// ---------------------------------------------------------------------------
__global__ __launch_bounds__(256) void flash_attn(
    const half_t* __restrict__ qbuf,   // [B*N, 512], pre-scaled by 1/64
    const half_t* __restrict__ kvbuf,  // [B*N, 1024] (k | v)
    const half_t* __restrict__ vtbuf,  // [(b*H+h)*64 + dh][N]
    half_t* __restrict__ obuf,         // [B*N, 512]
    int Nn, int Hh, int Dh)
{
  __shared__ half_t Ks[2][64][72];
  __shared__ half_t VsT[2][80][72];    // rows 64..79: ones row + zeros
  __shared__ half_t Ps[8][16][72];

  const int tid  = threadIdx.x;
  const int wave = tid >> 5;
  const int lane = tid & 31;
  const int b = blockIdx.z, h = blockIdx.y;
  const int q0_blk = blockIdx.x * 128;
  const int q0 = q0_blk + wave * 16;
  const int HD  = Hh * Dh;     // 512
  const int KVW = 2 * HD;      // 1024

  // init ones/zero rows of both V^T buffers (read only after first barrier)
  for (int idx = tid; idx < 2 * 16 * 72; idx += 256) {
    const int pb2 = idx / (16 * 72);
    const int rem = idx % (16 * 72);
    VsT[pb2][64 + rem / 72][rem % 72] = (rem / 72 == 0) ? (half_t)1.0f : (half_t)0.0f;
  }

  // Q fragments (16x64 -> two A-frags)
  const half_t* Qb = qbuf + (size_t)(b * Nn + q0) * HD + h * Dh;
  v16h aq[2];
  {
    const int m = lane & 15, hf = lane >> 4;
#pragma unroll
    for (int t = 0; t < 2; ++t) {
      v16h f;
#pragma unroll
      for (int i = 0; i < 8; ++i) {
        f[i]     = Qb[(size_t)m * HD + t * 32 + hf * 16 + i];
        f[8 + i] = Qb[(size_t)m * HD + t * 32 + hf * 16 + 8 + i];
      }
      aq[t] = f;
    }
  }

  float mrow[8];
#pragma unroll
  for (int r = 0; r < 8; ++r) mrow[r] = -3.0e38f;
  v8f accO[4] = {};
  v8f accL = {};

  const half_t* Kbase  = kvbuf + (size_t)(b * Nn) * KVW + h * Dh;
  const half_t* VTbase = vtbuf + (size_t)(b * Hh + h) * Dh * Nn;
  const int kend = q0_blk + 128;

  const int kr = tid >> 2, kc = (tid & 3) * 16;  // 64 x 64 tiles
  auto issue = [&](int c, int pb2) {
    const half_t* ks = Kbase + (size_t)(c + kr) * KVW + kc;
    async_copy_b128(&Ks[pb2][kr][kc],      ks);
    async_copy_b128(&Ks[pb2][kr][kc + 8],  ks + 8);
    const half_t* vs = VTbase + (size_t)kr * Nn + c + kc;
    async_copy_b128(&VsT[pb2][kr][kc],     vs);
    async_copy_b128(&VsT[pb2][kr][kc + 8], vs + 8);
  };

  issue(0, 0);
  int pb = 0;
  for (int c = 0; c < kend; c += 64, pb ^= 1) {
    wait_async0();
    __syncthreads();
    if (c + 64 < kend) issue(c + 64, pb ^ 1);

    if (c + 63 <= q0)
      attn_chunk<false>(aq, mrow, accO, accL, Ks[pb], VsT[pb], Ps[wave], q0, c);
    else if (c <= q0 + 15)
      attn_chunk<true>(aq, mrow, accO, accL, Ks[pb], VsT[pb], Ps[wave], q0, c);
  }

  // finalize: broadcast l (column 0 of accL) from lane hf*16, divide, store
  {
    const int n = lane & 15, hf = lane >> 4;
    float linv[8];
#pragma unroll
    for (int r = 0; r < 8; ++r) {
      const float l = __shfl(accL[r], lane & 16, 32);
      linv[r] = 1.0f / l;
    }
    half_t* Ob = obuf + (size_t)(b * Nn + q0) * HD + h * Dh;
#pragma unroll
    for (int t = 0; t < 4; ++t)
#pragma unroll
      for (int r = 0; r < 8; ++r)
        Ob[(size_t)(r + 8 * hf) * HD + t * 16 + n] = (half_t)(accO[t][r] * linv[r]);
  }
}

// ---------------------------------------------------------------------------
extern "C" void kernel_launch(void* const* d_in, const int* in_sizes, int n_in,
                              void* d_out, int out_size, void* d_ws, size_t ws_size,
                              hipStream_t stream) {
  const float* x   = (const float*)d_in[0];
  const float* Wq  = (const float*)d_in[1];
  const float* Wkv = (const float*)d_in[2];
  const float* Wo  = (const float*)d_in[3];
  const float* bo  = (const float*)d_in[4];
  float* out = (float*)d_out;

  const int Bc = 2, Nc = 4096, DIMc = 512, Hc = 8, DHc = 64;
  const int M  = Bc * Nc;       // 8192
  const int HD = Hc * DHc;      // 512

  half_t* ws = (half_t*)d_ws;
  half_t* xh    = ws;                           // M*512
  half_t* wqT   = xh   + (size_t)M * DIMc;      // [512][512]
  half_t* wkvT  = wqT  + (size_t)DIMc * HD;     // [1024][512]
  half_t* woT   = wkvT + (size_t)DIMc * 2 * HD; // [512][512]
  half_t* qbuf  = woT  + (size_t)HD * DIMc;     // M*512
  half_t* kvbuf = qbuf + (size_t)M * HD;        // M*1024
  half_t* vtbuf = kvbuf + (size_t)M * 2 * HD;   // M*512
  half_t* obuf  = vtbuf + (size_t)M * HD;       // M*512

  dim3 blk(256);

  convert_f32_f16<<<dim3((M * DIMc) / (256 * 8)), blk, 0, stream>>>(x, xh, M * DIMc);
  wt_transpose<<<dim3(HD / 64, DIMc / 64), blk, 0, stream>>>(Wq, wqT, DIMc, HD);
  wt_transpose<<<dim3(2 * HD / 64, DIMc / 64), blk, 0, stream>>>(Wkv, wkvT, DIMc, 2 * HD);
  wt_transpose<<<dim3(DIMc / 64, HD / 64), blk, 0, stream>>>(Wo, woT, HD, DIMc);

  // Q = (x @ Wq) * DH^-1  (reference applies DH^-0.5 twice)
  gemm_wmma<half_t, false><<<dim3(HD / 64, M / 128), blk, 0, stream>>>(
      xh, wqT, nullptr, qbuf, M, HD, DIMc, 1.0f / 64.0f);
  gemm_wmma<half_t, false><<<dim3(2 * HD / 64, M / 128), blk, 0, stream>>>(
      xh, wkvT, nullptr, kvbuf, M, 2 * HD, DIMc, 1.0f);

  v_transpose<<<dim3(Nc / 64, Hc, Bc), blk, 0, stream>>>(kvbuf, vtbuf, Nc);

  flash_attn<<<dim3(Nc / 128, Hc, Bc), blk, 0, stream>>>(
      qbuf, kvbuf, vtbuf, obuf, Nc, Hc, DHc);

  gemm_wmma<float, true><<<dim3(DIMc / 64, M / 128), blk, 0, stream>>>(
      obuf, woT, bo, out, M, DIMc, HD, 1.0f);
}
